// HyperJetNet_79379585565501
// MI455X (gfx1250) — matrix operand order repaired and checked by reading, not compiled, not run on hardware
//
#include <hip/hip_runtime.h>

#define B_  256
#define N_  512
#define E_  512
#define M_  640
#define D_  128
#define G_  384   // 3*D

typedef __bf16 bf16_t;
typedef bf16_t v16bf __attribute__((ext_vector_type(16)));
typedef float  v8f   __attribute__((ext_vector_type(8)));
typedef int    v4i_  __attribute__((ext_vector_type(4)));

__device__ __forceinline__ unsigned short f2bf(float x) {
  union { float f; unsigned u; } v; v.f = x;
  unsigned r = v.u + 0x7FFFu + ((v.u >> 16) & 1u);   // round-to-nearest-even
  return (unsigned short)(r >> 16);
}

__device__ __forceinline__ v8f wmma_bf16(v16bf a, v16bf b, v8f c) {
#if defined(__gfx1250__)
  // emits v_wmma_f32_16x16x32_bf16 (probe-confirmed signature)
  return __builtin_amdgcn_wmma_f32_16x16x32_bf16(false, a, false, b, (short)0, c, false, false);
#else
  return c;
#endif
}

// ---- async global->LDS (GLOBAL_LOAD_ASYNC_TO_LDS_B128, ASYNCcnt) ----
#if defined(__gfx1250__)
#if __has_builtin(__builtin_amdgcn_global_load_async_to_lds_b128) && \
    __has_builtin(__builtin_amdgcn_s_wait_asynccnt)
#define ASYNC_COPY_OK 1
#else
#define ASYNC_COPY_OK 0
#endif
#else
#define ASYNC_COPY_OK 0
#endif

__device__ __forceinline__ void copy16_g2l(unsigned short* ldst,
                                           const unsigned short* gsrc) {
#if ASYNC_COPY_OK
  __builtin_amdgcn_global_load_async_to_lds_b128(
      (__attribute__((address_space(1))) v4i_*)gsrc,
      (__attribute__((address_space(3))) v4i_*)ldst, 0, 0);
#else
  *(uint4*)ldst = *(const uint4*)gsrc;
#endif
}
__device__ __forceinline__ void async_wait() {
#if ASYNC_COPY_OK
  __builtin_amdgcn_s_wait_asynccnt(0);
#endif
}

// ---- fragment builders (wave32, contiguous k order, same for A and B), vectorized ----
// A 16x32 bf16: per lane K = {kb..kb+15}, kb=(lane>=16)*16.
// Both chunks are 16B-contiguous -> two b128 LDS loads. ld must be mult of 8.
__device__ __forceinline__ v16bf load_A_frag(const unsigned short* lds, int ld, int lane) {
  const int m = lane & 15, kb = (lane >> 4) * 16;
  const unsigned short* r0 = lds + m * ld + kb;
  union { v16bf v; uint4 q[2]; } f;
  f.q[0] = *(const uint4*)(r0);
  f.q[1] = *(const uint4*)(r0 + 8);
  return f.v;
}
// Same K pattern, source is f32 LDS (GRU hidden state), convert on the fly.
__device__ __forceinline__ v16bf load_A_frag_f32(const float* lds, int ld, int lane) {
  const int m = lane & 15, kb = (lane >> 4) * 16;
  const float* r0 = lds + m * ld + kb;
  union { v16bf v; unsigned short u[16]; } f;
#pragma unroll
  for (int e = 0; e < 8; ++e) f.u[e] = f2bf(r0[e]);
#pragma unroll
  for (int e = 0; e < 8; ++e) f.u[8 + e] = f2bf(r0[8 + e]);
  return f.v;
}
// B 32x16 bf16 from an N-major LDS tile (tile[n][k]): per lane N=lane&15,
// K = {kb..kb+15}, kb=(lane>=16)*16 -> two b128 LDS loads.
__device__ __forceinline__ v16bf load_B_fragT(const unsigned short* lds, int ld, int lane) {
  const int n = lane & 15, kb = (lane >> 4) * 16;
  const unsigned short* r0 = lds + n * ld + kb;
  union { v16bf v; uint4 q[2]; } f;
  f.q[0] = *(const uint4*)(r0);
  f.q[1] = *(const uint4*)(r0 + 8);
  return f.v;
}

// =====================================================================
// Stage obs -> Hbf (B,N,E) bf16, Xbf (B,N,D) bf16, dvis = 1/(sqrt(dv)+eps)
__global__ __launch_bounds__(128) void k_stage_obs(
    const float* __restrict__ obs, unsigned short* __restrict__ Hbf,
    unsigned short* __restrict__ Xbf, float* __restrict__ dvis) {
  __shared__ float wsum[4];
  const int n = blockIdx.x, b = blockIdx.y;
  const int tid = threadIdx.x, lane = tid & 31, wave = tid >> 5;
  const float* row = obs + ((long long)b * N_ + n) * M_;
  float4 h4 = *(const float4*)(row + tid * 4);
  ushort4 hv;
  hv.x = f2bf(h4.x); hv.y = f2bf(h4.y); hv.z = f2bf(h4.z); hv.w = f2bf(h4.w);
  *(ushort4*)&Hbf[((long long)b * N_ + n) * E_ + tid * 4] = hv;
  if (tid < 32) {
    float4 x4 = *(const float4*)(row + 512 + tid * 4);
    ushort4 xv;
    xv.x = f2bf(x4.x); xv.y = f2bf(x4.y); xv.z = f2bf(x4.z); xv.w = f2bf(x4.w);
    *(ushort4*)&Xbf[((long long)b * N_ + n) * D_ + tid * 4] = xv;
  }
  float s = h4.x + h4.y + h4.z + h4.w;
#pragma unroll
  for (int off = 16; off; off >>= 1) s += __shfl_xor(s, off, 32);
  if (lane == 0) wsum[wave] = s;
  __syncthreads();
  if (tid == 0)
    dvis[(long long)b * N_ + n] =
        1.f / (sqrtf(wsum[0] + wsum[1] + wsum[2] + wsum[3]) + 1e-8f);
}

// edge degrees: deiwe{1,2}[b][e] = w{1,2}_edge[e] / (d_e + eps)
__global__ __launch_bounds__(256) void k_dege(
    const float* __restrict__ obs, const float* __restrict__ w1,
    const float* __restrict__ w2, float* __restrict__ deiwe1,
    float* __restrict__ deiwe2) {
  const int b = blockIdx.x, tid = threadIdx.x;
  const int e = tid * 2;
  const float* base = obs + (long long)b * N_ * M_ + e;
  float sx = 0.f, sy = 0.f;
  for (int n = 0; n < N_; ++n) {
    float2 v = *(const float2*)(base + (long long)n * M_);
    sx += v.x; sy += v.y;
  }
  const long long o = (long long)b * E_ + e;
  float d0 = 1.f / (sx + 1e-8f), d1 = 1.f / (sy + 1e-8f);
  deiwe1[o]     = w1[e] * d0;
  deiwe1[o + 1] = w1[e + 1] * d1;
  deiwe2[o]     = w2[e] * d0;
  deiwe2[o + 1] = w2[e + 1] * d1;
}

// f32 -> bf16, layout preserved (count4 = elements/4)
__global__ void k_convert_bf(const float* __restrict__ in,
                             unsigned short* __restrict__ out, int count4) {
  int i = blockIdx.x * blockDim.x + threadIdx.x;
  if (i < count4) {
    float4 v = ((const float4*)in)[i];
    ushort4 o;
    o.x = f2bf(v.x); o.y = f2bf(v.y); o.z = f2bf(v.z); o.w = f2bf(v.w);
    ((ushort4*)out)[i] = o;
  }
}

// =====================================================================
// Batched WMMA GEMM, 128x64 tile, BK=32, 8 waves x 4 WMMA per K-step.
// TA=0 -> A[m][k]; TA=1 -> A[k][m].  TB=0 -> B[k][n]; TB=1 -> B[n][k].
// epilogue: v = rowscale[m] * (acc + colbias[n]); optional relu; bf16/f32 out
template <int TA, int TB>
__global__ __launch_bounds__(256) void k_gemm(
    const unsigned short* __restrict__ A, long long strideAb, int lda,
    const unsigned short* __restrict__ Bg, long long strideBb, int ldb,
    const float* __restrict__ rowscale, long long strideRSb,
    const float* __restrict__ colbias,
    unsigned short* __restrict__ outBf, long long strideObf,
    float* __restrict__ outF32, long long strideOf,
    int M, int Nc, int K, int relu) {
  __shared__ unsigned short Asm[128][40];  // [m][k], stride 40 (mult of 8)
  __shared__ unsigned short Bsm[64][40];   // [n][k], stride 40
  const int b = blockIdx.z;
  const int m0 = blockIdx.y * 128;
  const int n0 = blockIdx.x * 64;
  const unsigned short* Ab = A + (long long)b * strideAb;
  const unsigned short* Bb = Bg + (long long)b * strideBb;
  const int tid = threadIdx.x, lane = tid & 31, wave = tid >> 5;
  const int wm = (wave & 3) * 32;
  const int wn = (wave >> 2) * 32;
  // hoisted staging coordinates
  const int ar0 = tid >> 2, ac0 = (tid & 3) * 8;   // TA==0: rows ar0, ar0+64
  const int akk = tid >> 4, amm = (tid & 15) * 8;  // TA==1: kk akk, akk+16
  const int bkk = tid >> 3, bnn = (tid & 7) * 8;   // TB==0
  const int bn0 = tid >> 2, bc0 = (tid & 3) * 8;   // TB==1
  v8f acc00 = {}, acc01 = {}, acc10 = {}, acc11 = {};
  for (int k0 = 0; k0 < K; k0 += 32) {
    if (TA == 0) {
      copy16_g2l(&Asm[ar0][ac0],
                 &Ab[(long long)(m0 + ar0) * lda + k0 + ac0]);
      copy16_g2l(&Asm[ar0 + 64][ac0],
                 &Ab[(long long)(m0 + ar0 + 64) * lda + k0 + ac0]);
    } else {
#pragma unroll
      for (int i = 0; i < 2; ++i) {
        int kk = akk + i * 16;          // read contiguous m, scatter to [m][k]
        uint4 q = *(const uint4*)&Ab[(long long)(k0 + kk) * lda + m0 + amm];
        unsigned w0 = q.x, w1 = q.y, w2 = q.z, w3 = q.w;
        Asm[amm + 0][kk] = (unsigned short)(w0 & 0xffff);
        Asm[amm + 1][kk] = (unsigned short)(w0 >> 16);
        Asm[amm + 2][kk] = (unsigned short)(w1 & 0xffff);
        Asm[amm + 3][kk] = (unsigned short)(w1 >> 16);
        Asm[amm + 4][kk] = (unsigned short)(w2 & 0xffff);
        Asm[amm + 5][kk] = (unsigned short)(w2 >> 16);
        Asm[amm + 6][kk] = (unsigned short)(w3 & 0xffff);
        Asm[amm + 7][kk] = (unsigned short)(w3 >> 16);
      }
    }
    if (TB == 0) {                      // read contiguous n, scatter to [n][k]
      uint4 q = *(const uint4*)&Bb[(long long)(k0 + bkk) * ldb + n0 + bnn];
      unsigned w0 = q.x, w1 = q.y, w2 = q.z, w3 = q.w;
      Bsm[bnn + 0][bkk] = (unsigned short)(w0 & 0xffff);
      Bsm[bnn + 1][bkk] = (unsigned short)(w0 >> 16);
      Bsm[bnn + 2][bkk] = (unsigned short)(w1 & 0xffff);
      Bsm[bnn + 3][bkk] = (unsigned short)(w1 >> 16);
      Bsm[bnn + 4][bkk] = (unsigned short)(w2 & 0xffff);
      Bsm[bnn + 5][bkk] = (unsigned short)(w2 >> 16);
      Bsm[bnn + 6][bkk] = (unsigned short)(w3 & 0xffff);
      Bsm[bnn + 7][bkk] = (unsigned short)(w3 >> 16);
    } else {                            // contiguous k both sides
      copy16_g2l(&Bsm[bn0][bc0],
                 &Bb[(long long)(n0 + bn0) * ldb + k0 + bc0]);
    }
    async_wait();
    __syncthreads();
    v16bf a0 = load_A_frag(&Asm[wm][0], 40, lane);
    v16bf a1 = load_A_frag(&Asm[wm + 16][0], 40, lane);
    v16bf b0 = load_B_fragT(&Bsm[wn][0], 40, lane);
    v16bf b1 = load_B_fragT(&Bsm[wn + 16][0], 40, lane);
    acc00 = wmma_bf16(a0, b0, acc00);
    acc01 = wmma_bf16(a0, b1, acc01);
    acc10 = wmma_bf16(a1, b0, acc10);
    acc11 = wmma_bf16(a1, b1, acc11);
    __syncthreads();
  }
  const int lane16 = lane & 15, lh = lane >> 4;
  const float* rs = rowscale ? rowscale + (long long)b * strideRSb : nullptr;
#pragma unroll
  for (int mi = 0; mi < 2; ++mi) {
#pragma unroll
    for (int ni = 0; ni < 2; ++ni) {
      v8f a = mi ? (ni ? acc11 : acc10) : (ni ? acc01 : acc00);
#pragma unroll
      for (int r = 0; r < 8; ++r) {
        int m = m0 + wm + mi * 16 + lh * 8 + r;
        int n = n0 + wn + ni * 16 + lane16;
        float v = a[r];
        if (colbias) v += colbias[n];
        if (rs) v *= rs[m];
        if (relu) v = fmaxf(v, 0.f);
        if (outBf)  outBf [(long long)b * strideObf + (long long)m * Nc + n] = f2bf(v);
        if (outF32) outF32[(long long)b * strideOf  + (long long)m * Nc + n] = v;
      }
    }
  }
}

// =====================================================================
__global__ __launch_bounds__(128) void k_layernorm(
    const float* __restrict__ Zf, const float* __restrict__ g,
    const float* __restrict__ bb, float* __restrict__ Znf,
    unsigned short* __restrict__ Znbf) {
  __shared__ float rsum[4], rsq[4];
  const int n = blockIdx.x, b = blockIdx.y;
  const int tid = threadIdx.x, lane = tid & 31, wave = tid >> 5;
  const long long row = ((long long)b * N_ + n) * D_;
  float x = Zf[row + tid];
  float s = x, q = x * x;
#pragma unroll
  for (int off = 16; off; off >>= 1) {
    s += __shfl_xor(s, off, 32);
    q += __shfl_xor(q, off, 32);
  }
  if (lane == 0) { rsum[wave] = s; rsq[wave] = q; }
  __syncthreads();
  float mu  = (rsum[0] + rsum[1] + rsum[2] + rsum[3]) * (1.f / 128.f);
  float var = (rsq[0] + rsq[1] + rsq[2] + rsq[3]) * (1.f / 128.f) - mu * mu;
  float y = (x - mu) * rsqrtf(var + 1e-5f) * g[tid] + bb[tid];
  Znf[row + tid]  = y;
  Znbf[row + tid] = f2bf(y);
}

// =====================================================================
// GRU encoder scan: 16 batches/WG; gates_x and gh both WMMA each step.
// Weights stay in native (3D, D) layout -> B-frag = two b128 global loads.
__global__ __launch_bounds__(256) void k_encoder(
    const unsigned short* __restrict__ Znbf,
    const unsigned short* __restrict__ wihB,   // (384 x 128) bf16, native
    const unsigned short* __restrict__ whhB,   // (384 x 128) bf16, native
    const float* __restrict__ b_ih, const float* __restrict__ b_hh,
    float* __restrict__ enc_outs, float* __restrict__ h_n) {
  __shared__ float h_lds[16][128];
  __shared__ float gx_lds[16][G_];
  __shared__ float gh_lds[16][G_];
  __shared__ unsigned short ztile[16][136];
  const int tid = threadIdx.x, lane = tid & 31, wave = tid >> 5;
  const int b0 = blockIdx.x * 16;
  const int zm = tid >> 4, zd = (tid & 15) * 8;   // ztile staging coords
  const unsigned short* zsrc = Znbf + ((long long)(b0 + zm) * N_) * D_ + zd;
  for (int i = tid; i < 16 * 128; i += 256) h_lds[i >> 7][i & 127] = 0.f;
  __syncthreads();
  for (int n = 0; n < N_; ++n) {
    copy16_g2l(&ztile[zm][zd], zsrc + (long long)n * D_);
    async_wait();
    __syncthreads();
    for (int ct = wave; ct < 24; ct += 8) {
      const int g0 = ct * 16;
      const int gcol = g0 + (lane & 15);
      const unsigned short* wxr = wihB + (long long)gcol * D_;
      const unsigned short* whr = whhB + (long long)gcol * D_;
      v8f ax = {}; v8f ah = {};
#pragma unroll
      for (int kc = 0; kc < 4; ++kc) {
        const int k0 = kc * 32;
        v16bf aZ = load_A_frag(&ztile[0][k0], 136, lane);
        v16bf aH = load_A_frag_f32(&h_lds[0][k0], 128, lane);
        const int kb2 = k0 + (lane >> 4) * 16;
        union { v16bf v; uint4 q[2]; } fx, fh;
        fx.q[0] = *(const uint4*)(wxr + kb2);
        fx.q[1] = *(const uint4*)(wxr + kb2 + 8);
        fh.q[0] = *(const uint4*)(whr + kb2);
        fh.q[1] = *(const uint4*)(whr + kb2 + 8);
        ax = wmma_bf16(aZ, fx.v, ax);
        ah = wmma_bf16(aH, fh.v, ah);
      }
      const int mrow = (lane >> 4) * 8;
      const float bx = b_ih[gcol], bh = b_hh[gcol];
#pragma unroll
      for (int r = 0; r < 8; ++r) {
        gx_lds[mrow + r][gcol] = ax[r] + bx;
        gh_lds[mrow + r][gcol] = ah[r] + bh;
      }
    }
    __syncthreads();
    for (int i = tid; i < 16 * 128; i += 256) {
      int m = i >> 7, d = i & 127;
      float r  = 1.f / (1.f + expf(-(gx_lds[m][d]       + gh_lds[m][d])));
      float z  = 1.f / (1.f + expf(-(gx_lds[m][d + 128] + gh_lds[m][d + 128])));
      float nn = tanhf(gx_lds[m][d + 256] + r * gh_lds[m][d + 256]);
      float hnew = (1.f - z) * nn + z * h_lds[m][d];
      h_lds[m][d] = hnew;
      enc_outs[((long long)(b0 + m) * N_ + n) * D_ + d] = hnew;
    }
    __syncthreads();
  }
  for (int i = tid; i < 16 * 128; i += 256)
    h_n[(long long)(b0 + (i >> 7)) * D_ + (i & 127)] = h_lds[i >> 7][i & 127];
}

// =====================================================================
// attention + decoder GRU head (tiny): one WG of 128 threads per batch
__global__ __launch_bounds__(128) void k_head(
    const float* __restrict__ h_n, const float* __restrict__ enc_outs,
    const float* __restrict__ Znf, const float* __restrict__ attn_w,
    const float* __restrict__ dec_w_ih, const float* __restrict__ dec_w_hh,
    const float* __restrict__ dec_b_ih, const float* __restrict__ dec_b_hh,
    const int* __restrict__ task_idx, float* __restrict__ out) {
  __shared__ float hid[128], q[128], ctx[128], cur[128];
  __shared__ float sc[512];
  __shared__ float gx[384], gh[384];
  __shared__ float red[4];
  const int b = blockIdx.x, tid = threadIdx.x, lane = tid & 31, wave = tid >> 5;
  hid[tid] = h_n[(long long)b * 128 + tid];
  const int task = task_idx[0];
  cur[tid] = Znf[((long long)b * N_ + task) * D_ + tid];
  __syncthreads();
  // query = hidden @ attn_w.T
  float s = 0.f;
  for (int k = 0; k < 128; ++k) s += hid[k] * attn_w[tid * 128 + k];
  q[tid] = s;
  __syncthreads();
  // scores
  const float inv = 0.08838834764831845f;  // 1/sqrt(128)
  for (int n = wave; n < 512; n += 4) {
    float p = 0.f;
    const float* eo = enc_outs + ((long long)b * N_ + n) * D_;
    for (int k = lane; k < 128; k += 32) p += q[k] * eo[k];
#pragma unroll
    for (int off = 16; off; off >>= 1) p += __shfl_xor(p, off, 32);
    if (lane == 0) sc[n] = p * inv;
  }
  __syncthreads();
  // softmax
  float mx = -3.4e38f;
  for (int n = tid; n < 512; n += 128) mx = fmaxf(mx, sc[n]);
#pragma unroll
  for (int off = 16; off; off >>= 1) mx = fmaxf(mx, __shfl_xor(mx, off, 32));
  if (lane == 0) red[wave] = mx;
  __syncthreads();
  mx = fmaxf(fmaxf(red[0], red[1]), fmaxf(red[2], red[3]));
  float ssum = 0.f;
  for (int n = tid; n < 512; n += 128) { float e = expf(sc[n] - mx); sc[n] = e; ssum += e; }
#pragma unroll
  for (int off = 16; off; off >>= 1) ssum += __shfl_xor(ssum, off, 32);
  __syncthreads();
  if (lane == 0) red[wave] = ssum;
  __syncthreads();
  float rinv = 1.f / (red[0] + red[1] + red[2] + red[3]);
  // context
  float c = 0.f;
  for (int n = 0; n < 512; ++n) c += sc[n] * enc_outs[((long long)b * N_ + n) * D_ + tid];
  ctx[tid] = c * rinv;
  __syncthreads();
  // decoder GEMVs
  for (int g = tid; g < 384; g += 128) {
    float a = dec_b_ih[g];
    const float* w = dec_w_ih + (long long)g * 256;
    for (int k = 0; k < 128; ++k) a += ctx[k] * w[k];
    for (int k = 0; k < 128; ++k) a += cur[k] * w[128 + k];
    gx[g] = a;
    float hs = dec_b_hh[g];
    const float* wh = dec_w_hh + (long long)g * 128;
    for (int k = 0; k < 128; ++k) hs += hid[k] * wh[k];
    gh[g] = hs;
  }
  __syncthreads();
  {
    int d = tid;
    float r  = 1.f / (1.f + expf(-(gx[d] + gh[d])));
    float z  = 1.f / (1.f + expf(-(gx[d + 128] + gh[d + 128])));
    float nn = tanhf(gx[d + 256] + r * gh[d + 256]);
    out[(long long)b * 128 + d] = (1.f - z) * nn + z * hid[d];
  }
}

// =====================================================================
extern "C" void kernel_launch(void* const* d_in, const int* in_sizes, int n_in,
                              void* d_out, int out_size, void* d_ws, size_t ws_size,
                              hipStream_t stream) {
  (void)in_sizes; (void)n_in; (void)out_size; (void)ws_size;
  const float* obs  = (const float*)d_in[0];
  const float* t1w  = (const float*)d_in[1];
  const float* t1b  = (const float*)d_in[2];
  const float* w1e  = (const float*)d_in[3];
  const float* t2w  = (const float*)d_in[4];
  const float* t2b  = (const float*)d_in[5];
  const float* w2e  = (const float*)d_in[6];
  const float* lng  = (const float*)d_in[7];
  const float* lnb  = (const float*)d_in[8];
  const float* wih  = (const float*)d_in[9];
  const float* whh  = (const float*)d_in[10];
  const float* bih  = (const float*)d_in[11];
  const float* bhh  = (const float*)d_in[12];
  const float* attw = (const float*)d_in[13];
  const float* dwih = (const float*)d_in[14];
  const float* dwhh = (const float*)d_in[15];
  const float* dbih = (const float*)d_in[16];
  const float* dbhh = (const float*)d_in[17];
  const int*   task = (const int*)d_in[18];
  float* out = (float*)d_out;

  char* p = (char*)d_ws;
  auto alloc = [&](size_t bytes) -> void* {
    void* r = (void*)p;
    p += (bytes + 255) & ~(size_t)255;
    return r;
  };
  float* dvis   = (float*)alloc((size_t)B_ * N_ * 4);
  float* deiwe1 = (float*)alloc((size_t)B_ * E_ * 4);
  float* deiwe2 = (float*)alloc((size_t)B_ * E_ * 4);
  unsigned short* Hbf  = (unsigned short*)alloc((size_t)B_ * N_ * E_ * 2);
  unsigned short* Xbf  = (unsigned short*)alloc((size_t)B_ * N_ * D_ * 2);
  unsigned short* XsBf = (unsigned short*)alloc((size_t)B_ * N_ * D_ * 2);
  unsigned short* EwBf = (unsigned short*)alloc((size_t)B_ * E_ * D_ * 2);
  unsigned short* Z1Bf = (unsigned short*)alloc((size_t)B_ * N_ * D_ * 2);
  float* Z2f  = (float*)alloc((size_t)B_ * N_ * D_ * 4);
  float* Znf  = (float*)alloc((size_t)B_ * N_ * D_ * 4);
  unsigned short* ZnBf = (unsigned short*)alloc((size_t)B_ * N_ * D_ * 2);
  unsigned short* t1wB = (unsigned short*)alloc((size_t)D_ * D_ * 2);
  unsigned short* t2wB = (unsigned short*)alloc((size_t)D_ * D_ * 2);
  unsigned short* wihB = (unsigned short*)alloc((size_t)G_ * D_ * 2);
  unsigned short* whhB = (unsigned short*)alloc((size_t)G_ * D_ * 2);
  float* enc_outs = (float*)alloc((size_t)B_ * N_ * D_ * 4);
  float* hN       = (float*)alloc((size_t)B_ * D_ * 4);

  k_stage_obs<<<dim3(N_, B_), 128, 0, stream>>>(obs, Hbf, Xbf, dvis);
  k_dege<<<B_, 256, 0, stream>>>(obs, w1e, w2e, deiwe1, deiwe2);
  k_convert_bf<<<(D_ * D_ / 4 + 255) / 256, 256, 0, stream>>>(t1w, t1wB, D_ * D_ / 4);
  k_convert_bf<<<(D_ * D_ / 4 + 255) / 256, 256, 0, stream>>>(t2w, t2wB, D_ * D_ / 4);
  k_convert_bf<<<(G_ * D_ / 4 + 255) / 256, 256, 0, stream>>>(wih, wihB, G_ * D_ / 4);
  k_convert_bf<<<(G_ * D_ / 4 + 255) / 256, 256, 0, stream>>>(whh, whhB, G_ * D_ / 4);

  // ---- layer 1 ----
  // Xs = dvis * (X @ t1w.T + t1b)      (B operand in native (n,k) layout)
  k_gemm<0, 1><<<dim3(D_ / 64, N_ / 128, B_), 256, 0, stream>>>(
      Xbf, (long long)N_ * D_, D_, t1wB, 0, D_,
      dvis, N_, t1b, XsBf, (long long)N_ * D_, nullptr, 0, N_, D_, D_, 0);
  // Ew = (we*dei) * (H^T @ Xs)
  k_gemm<1, 0><<<dim3(D_ / 64, E_ / 128, B_), 256, 0, stream>>>(
      Hbf, (long long)N_ * E_, E_, XsBf, (long long)N_ * D_, D_,
      deiwe1, E_, nullptr, EwBf, (long long)E_ * D_, nullptr, 0, E_, D_, N_, 0);
  // Z1 = relu(dvis * (H @ Ew))
  k_gemm<0, 0><<<dim3(D_ / 64, N_ / 128, B_), 256, 0, stream>>>(
      Hbf, (long long)N_ * E_, E_, EwBf, (long long)E_ * D_, D_,
      dvis, N_, nullptr, Z1Bf, (long long)N_ * D_, nullptr, 0, N_, D_, E_, 1);
  // ---- layer 2 ----
  k_gemm<0, 1><<<dim3(D_ / 64, N_ / 128, B_), 256, 0, stream>>>(
      Z1Bf, (long long)N_ * D_, D_, t2wB, 0, D_,
      dvis, N_, t2b, XsBf, (long long)N_ * D_, nullptr, 0, N_, D_, D_, 0);
  k_gemm<1, 0><<<dim3(D_ / 64, E_ / 128, B_), 256, 0, stream>>>(
      Hbf, (long long)N_ * E_, E_, XsBf, (long long)N_ * D_, D_,
      deiwe2, E_, nullptr, EwBf, (long long)E_ * D_, nullptr, 0, E_, D_, N_, 0);
  k_gemm<0, 0><<<dim3(D_ / 64, N_ / 128, B_), 256, 0, stream>>>(
      Hbf, (long long)N_ * E_, E_, EwBf, (long long)E_ * D_, D_,
      dvis, N_, nullptr, nullptr, 0, Z2f, (long long)N_ * D_, N_, D_, E_, 1);

  k_layernorm<<<dim3(N_, B_), 128, 0, stream>>>(Z2f, lng, lnb, Znf, ZnBf);
  k_encoder<<<B_ / 16, 256, 0, stream>>>(ZnBf, wihB, whhB, bih, bhh, enc_outs, hN);
  k_head<<<B_, 128, 0, stream>>>(hN, enc_outs, Znf, attw, dwih, dwhh, dbih, dbhh, task, out);
}
